// MultiHeadAttention_35003983463195
// MI455X (gfx1250) — matrix operand order repaired and checked by reading, not compiled, not run on hardware
//
#include <hip/hip_runtime.h>
#include <hip/hip_bf16.h>

// ---------------------------------------------------------------------------
// MultiHeadAttention for MI455X (gfx1250), bf16 WMMA path, flash-attention.
// E=1024, H=16, D=64, B=2, S=4096.
// GEMMs: 32x64 output tile per wave (8 WMMA per 32-wide k-step).
// ---------------------------------------------------------------------------

typedef __bf16 bf16;
typedef __attribute__((ext_vector_type(8)))  __bf16 bf16x8;
typedef __attribute__((ext_vector_type(16))) __bf16 bf16x16;
typedef __attribute__((ext_vector_type(8)))  float  f32x8;

struct Frag {
    union {
        bf16x16 v;
        bf16x8  h[2];
    };
};

static __device__ inline f32x8 wmma_bf16(const Frag& a, const Frag& b, f32x8 c) {
    // D = A(16x32 bf16) * B(32x16 bf16) + C(16x16 f32)
    return __builtin_amdgcn_wmma_f32_16x16x32_bf16(
        /*neg_a=*/false, a.v, /*neg_b=*/false, b.v,
        /*c_mod=*/(short)0, c, /*reuse_a=*/false, /*reuse_b=*/false);
}

// -------------------------- elementwise converts ---------------------------

__global__ void k_cvt_bf16(const float* __restrict__ src, bf16* __restrict__ dst, int n) {
    int i = blockIdx.x * blockDim.x + threadIdx.x;
    if (i < n) dst[i] = (bf16)src[i];
}

// src: [K x N] row-major (f32)  ->  dst: [N x K] row-major (bf16)
__global__ void k_transpose_cvt(const float* __restrict__ src, bf16* __restrict__ dst,
                                int K, int N) {
    int i = blockIdx.x * blockDim.x + threadIdx.x;
    if (i < N * K) {
        int n = i / K;
        int k = i - n * K;
        dst[i] = (bf16)src[(size_t)k * N + n];
    }
}

// ------------------------------ QKV GEMM -----------------------------------
// C[M=8192, N=3072] = xb[M,1024] @ Wqkv + b ; scatter into Q, K (b,h,s,d) and
// Vt (b,h,d,s). Q pre-scaled by 1/sqrt(D)=0.125.
// Wave tile: 32 (M) x 64 (N).

__global__ __launch_bounds__(256) void k_qkv_gemm(
    const bf16* __restrict__ xb, const bf16* __restrict__ Wt,
    const float* __restrict__ bias,
    bf16* __restrict__ Qp, bf16* __restrict__ Kp, bf16* __restrict__ Vtp) {

    const int lane = threadIdx.x & 31;
    const int wid  = threadIdx.x >> 5;
    const int l16  = lane & 15;
    const int hi16 = lane >> 4;           // half: 0 or 1

    const int tid = blockIdx.x * 8 + wid;
    const int nt  = tid % 48;             // 3072/64
    const int mt  = tid / 48;             // 8192/32

    const bf16* arow0 = xb + (size_t)(mt * 32 + l16) * 1024;
    const bf16* arow1 = arow0 + (size_t)16 * 1024;
    const bf16* brow0 = Wt + (size_t)(nt * 64 + l16) * 1024 + 8 * hi16;

    f32x8 acc[2][4];
    #pragma unroll
    for (int mi = 0; mi < 2; ++mi)
        #pragma unroll
        for (int ni = 0; ni < 4; ++ni) acc[mi][ni] = f32x8{};

    for (int kk = 0; kk < 1024; kk += 32) {
        __builtin_prefetch(arow0 + kk + 256, 0, 1);
        __builtin_prefetch(brow0 + kk + 256, 0, 1);
        Frag a[2], b[4];
        a[0].h[0] = *(const bf16x8*)(arow0 + kk + 8 * hi16);
        a[0].h[1] = *(const bf16x8*)(arow0 + kk + 16 + 8 * hi16);
        a[1].h[0] = *(const bf16x8*)(arow1 + kk + 8 * hi16);
        a[1].h[1] = *(const bf16x8*)(arow1 + kk + 16 + 8 * hi16);
        #pragma unroll
        for (int ni = 0; ni < 4; ++ni) {
            const bf16* br = brow0 + (size_t)(16 * ni) * 1024 + kk;
            b[ni].h[0] = *(const bf16x8*)(br);
            b[ni].h[1] = *(const bf16x8*)(br + 16);
        }
        #pragma unroll
        for (int mi = 0; mi < 2; ++mi)
            #pragma unroll
            for (int ni = 0; ni < 4; ++ni)
                acc[mi][ni] = wmma_bf16(a[mi], b[ni], acc[mi][ni]);
    }

    #pragma unroll
    for (int ni = 0; ni < 4; ++ni) {
        const int n     = nt * 64 + 16 * ni + l16;
        const float bv  = bias[n];
        const int which = n >> 10;        // 0=Q 1=K 2=V
        const int e     = n & 1023;
        const int hh    = e >> 6;
        const int d     = e & 63;
        #pragma unroll
        for (int mi = 0; mi < 2; ++mi) {
            #pragma unroll
            for (int v = 0; v < 8; ++v) {
                const int mrow = mt * 32 + 16 * mi + v + 8 * hi16;
                const int bb   = mrow >> 12;   // / 4096
                const int s    = mrow & 4095;
                const float val = acc[mi][ni][v] + bv;
                const size_t bhs = (size_t)(bb * 16 + hh) * 4096 + s;
                if (which == 0)      Qp[bhs * 64 + d] = (bf16)(val * 0.125f);
                else if (which == 1) Kp[bhs * 64 + d] = (bf16)val;
                else                 Vtp[((size_t)(bb * 16 + hh) * 64 + d) * 4096 + s] = (bf16)val;
            }
        }
    }
}

// --------------------------- flash attention -------------------------------
// One wave handles one (b,h,16-query) tile; streams 32-key tiles.

__global__ __launch_bounds__(256) void k_attention(
    const bf16* __restrict__ Qp, const bf16* __restrict__ Kp,
    const bf16* __restrict__ Vtp, bf16* __restrict__ attn) {

    __shared__ bf16 Plds[8 * 16 * 32];    // 1KB per wave

    const int lane = threadIdx.x & 31;
    const int wid  = threadIdx.x >> 5;
    const int l16  = lane & 15;
    const int hi16 = lane >> 4;

    const int w  = blockIdx.x * 8 + wid;
    const int qt = w & 255;               // S/16
    const int hd = (w >> 8) & 15;
    const int b  = w >> 12;
    const int bh = b * 16 + hd;

    // Q fragments (held in registers across the whole loop), Q pre-scaled.
    const bf16* qrow = Qp + ((size_t)bh * 4096 + qt * 16 + l16) * 64;
    Frag qa[2];
    #pragma unroll
    for (int c = 0; c < 2; ++c) {
        qa[c].h[0] = *(const bf16x8*)(qrow + 32 * c + 16 * hi16);
        qa[c].h[1] = *(const bf16x8*)(qrow + 32 * c + 16 * hi16 + 8);
    }

    f32x8 O[4];
    float m[8], l[8];
    #pragma unroll
    for (int v = 0; v < 8; ++v) { m[v] = -1e30f; l[v] = 0.0f; }
    #pragma unroll
    for (int dt = 0; dt < 4; ++dt) O[dt] = f32x8{};

    bf16* Pw = Plds + wid * (16 * 32);
    const bf16* Kh = Kp  + (size_t)bh * 4096 * 64;
    const bf16* Vh = Vtp + (size_t)bh * 64 * 4096;

    for (int kt = 0; kt < 128; ++kt) {
        const int k0 = kt * 32;

        // ---- scores: two 16x16 tiles (keys k0..k0+15 and k0+16..k0+31) ----
        f32x8 s0 = {}, s1 = {};
        {
            const bf16* kr = Kh + (size_t)(k0 + l16) * 64 + 16 * hi16;
            Frag kb;
            kb.h[0] = *(const bf16x8*)(kr);      kb.h[1] = *(const bf16x8*)(kr + 8);
            s0 = wmma_bf16(qa[0], kb, s0);
            kb.h[0] = *(const bf16x8*)(kr + 32); kb.h[1] = *(const bf16x8*)(kr + 40);
            s0 = wmma_bf16(qa[1], kb, s0);
        }
        {
            const bf16* kr = Kh + (size_t)(k0 + 16 + l16) * 64 + 16 * hi16;
            Frag kb;
            kb.h[0] = *(const bf16x8*)(kr);      kb.h[1] = *(const bf16x8*)(kr + 8);
            s1 = wmma_bf16(qa[0], kb, s1);
            kb.h[0] = *(const bf16x8*)(kr + 32); kb.h[1] = *(const bf16x8*)(kr + 40);
            s1 = wmma_bf16(qa[1], kb, s1);
        }

        // ---- online softmax (rows live in vgpr index, cols across 16 lanes)
        float mt[8];
        #pragma unroll
        for (int v = 0; v < 8; ++v) mt[v] = fmaxf(s0[v], s1[v]);
        #pragma unroll
        for (int off = 1; off < 16; off <<= 1)
            #pragma unroll
            for (int v = 0; v < 8; ++v)
                mt[v] = fmaxf(mt[v], __shfl_xor(mt[v], off, 16));

        float corr[8], rs[8];
        #pragma unroll
        for (int v = 0; v < 8; ++v) {
            const float mn = fmaxf(m[v], mt[v]);
            corr[v] = __expf(m[v] - mn);
            m[v] = mn;
            const float p0 = __expf(s0[v] - mn);
            const float p1 = __expf(s1[v] - mn);
            rs[v] = p0 + p1;
            const int row = v + 8 * hi16;
            Pw[row * 32 + l16]      = (bf16)p0;
            Pw[row * 32 + 16 + l16] = (bf16)p1;
        }
        #pragma unroll
        for (int off = 1; off < 16; off <<= 1)
            #pragma unroll
            for (int v = 0; v < 8; ++v)
                rs[v] += __shfl_xor(rs[v], off, 16);
        #pragma unroll
        for (int v = 0; v < 8; ++v) {
            l[v] = l[v] * corr[v] + rs[v];
            O[0][v] *= corr[v]; O[1][v] *= corr[v];
            O[2][v] *= corr[v]; O[3][v] *= corr[v];
        }

        asm volatile("s_wait_dscnt 0" ::: "memory");

        // re-layout probs: C-frag (via LDS) -> A-frag
        Frag pa;
    pa.h[0] = *(const bf16x8*)(Pw + l16 * 32 + 16 * hi16);
    pa.h[1] = *(const bf16x8*)(Pw + l16 * 32 + 16 * hi16 + 8);

        // ---- O += P @ V  (V transposed: [d, s], contiguous over keys) ----
        #pragma unroll
        for (int dt = 0; dt < 4; ++dt) {
            const bf16* vr = Vh + (size_t)(16 * dt + l16) * 4096 + k0 + 16 * hi16;
            Frag vb;
            vb.h[0] = *(const bf16x8*)(vr);
            vb.h[1] = *(const bf16x8*)(vr + 8);
            O[dt] = wmma_bf16(pa, vb, O[dt]);
        }
    }

    // ---- normalize and store attn tile into [B*S, E] bf16 ----
    #pragma unroll
    for (int v = 0; v < 8; ++v) {
        const float inv = 1.0f / l[v];
        const size_t row = (size_t)b * 4096 + qt * 16 + v + 8 * hi16;
        #pragma unroll
        for (int dt = 0; dt < 4; ++dt) {
            const int col = hd * 64 + 16 * dt + l16;
            attn[row * 1024 + col] = (bf16)(O[dt][v] * inv);
        }
    }
}

// ------------------------------ out GEMM -----------------------------------
// out[8192,1024] f32 = attn[8192,1024] @ W_out + b_out
// Wave tile: 32 (M) x 64 (N).

__global__ __launch_bounds__(256) void k_out_gemm(
    const bf16* __restrict__ attn, const bf16* __restrict__ Wt,
    const float* __restrict__ bias, float* __restrict__ out) {

    const int lane = threadIdx.x & 31;
    const int wid  = threadIdx.x >> 5;
    const int l16  = lane & 15;
    const int hi16 = lane >> 4;

    const int tid = blockIdx.x * 8 + wid;
    const int nt  = tid & 15;             // 1024/64
    const int mt  = tid >> 4;             // 8192/32

    const bf16* arow0 = attn + (size_t)(mt * 32 + l16) * 1024;
    const bf16* arow1 = arow0 + (size_t)16 * 1024;
    const bf16* brow0 = Wt   + (size_t)(nt * 64 + l16) * 1024 + 8 * hi16;

    f32x8 acc[2][4];
    #pragma unroll
    for (int mi = 0; mi < 2; ++mi)
        #pragma unroll
        for (int ni = 0; ni < 4; ++ni) acc[mi][ni] = f32x8{};

    for (int kk = 0; kk < 1024; kk += 32) {
        __builtin_prefetch(arow0 + kk + 256, 0, 1);
        __builtin_prefetch(brow0 + kk + 256, 0, 1);
        Frag a[2], b[4];
        a[0].h[0] = *(const bf16x8*)(arow0 + kk + 8 * hi16);
        a[0].h[1] = *(const bf16x8*)(arow0 + kk + 16 + 8 * hi16);
        a[1].h[0] = *(const bf16x8*)(arow1 + kk + 8 * hi16);
        a[1].h[1] = *(const bf16x8*)(arow1 + kk + 16 + 8 * hi16);
        #pragma unroll
        for (int ni = 0; ni < 4; ++ni) {
            const bf16* br = brow0 + (size_t)(16 * ni) * 1024 + kk;
            b[ni].h[0] = *(const bf16x8*)(br);
            b[ni].h[1] = *(const bf16x8*)(br + 16);
        }
        #pragma unroll
        for (int mi = 0; mi < 2; ++mi)
            #pragma unroll
            for (int ni = 0; ni < 4; ++ni)
                acc[mi][ni] = wmma_bf16(a[mi], b[ni], acc[mi][ni]);
    }

    #pragma unroll
    for (int ni = 0; ni < 4; ++ni) {
        const int n    = nt * 64 + 16 * ni + l16;
        const float bv = bias[n];
        #pragma unroll
        for (int mi = 0; mi < 2; ++mi) {
            #pragma unroll
            for (int v = 0; v < 8; ++v) {
                const size_t mrow = (size_t)(mt * 32 + 16 * mi + v + 8 * hi16);
                out[mrow * 1024 + n] = acc[mi][ni][v] + bv;
            }
        }
    }
}

// ------------------------------- launcher ----------------------------------

extern "C" void kernel_launch(void* const* d_in, const int* in_sizes, int n_in,
                              void* d_out, int out_size, void* d_ws, size_t ws_size,
                              hipStream_t stream) {
    const float* x     = (const float*)d_in[0];   // [2,4096,1024]
    const float* W_qkv = (const float*)d_in[1];   // [1024,3072]
    const float* b_qkv = (const float*)d_in[2];   // [3072]
    const float* W_out = (const float*)d_in[3];   // [1024,1024]
    const float* b_out = (const float*)d_in[4];   // [1024]
    float* out = (float*)d_out;                   // [2,4096,1024]

    const size_t M   = 8192;                      // B*S
    const size_t E   = 1024;
    const size_t E3  = 3072;
    const size_t BHSD = 2ull * 16 * 4096 * 64;    // per Q/K/V buffer

    size_t off = 0;
    auto carve = [&](size_t bytes) {
        void* p = (char*)d_ws + off;
        off += (bytes + 255) & ~(size_t)255;
        return p;
    };
    bf16* xb     = (bf16*)carve(M * E * 2);
    bf16* Wqkv_t = (bf16*)carve(E3 * E * 2);
    bf16* Wout_t = (bf16*)carve(E * E * 2);
    bf16* Qp     = (bf16*)carve(BHSD * 2);
    bf16* Kp     = (bf16*)carve(BHSD * 2);
    bf16* Vtp    = (bf16*)carve(BHSD * 2);
    bf16* attn   = (bf16*)carve(M * E * 2);
    (void)ws_size; (void)in_sizes; (void)n_in; (void)out_size;

    // 1) converts / transposes
    k_cvt_bf16<<<(int)(M * E / 256), 256, 0, stream>>>(x, xb, (int)(M * E));
    k_transpose_cvt<<<(int)(E3 * E / 256), 256, 0, stream>>>(W_qkv, Wqkv_t, (int)E, (int)E3);
    k_transpose_cvt<<<(int)(E * E / 256), 256, 0, stream>>>(W_out, Wout_t, (int)E, (int)E);

    // 2) QKV projection: (8192/32)*(3072/64) = 256*48 wave tiles, 8 waves/block
    k_qkv_gemm<<<256 * 48 / 8, 256, 0, stream>>>(xb, Wqkv_t, b_qkv, Qp, Kp, Vtp);

    // 3) flash attention: 2*16*256 q-tiles, 8 waves/block
    k_attention<<<2 * 16 * 256 / 8, 256, 0, stream>>>(Qp, Kp, Vtp, attn);

    // 4) output projection: (8192/32)*(1024/64) = 256*16 wave tiles, 8 waves/block
    k_out_gemm<<<256 * 16 / 8, 256, 0, stream>>>(attn, Wout_t, b_out, out);
}
